// MinGRUcell_47605417509424
// MI455X (gfx1250) — matrix operand order repaired and checked by reading, not compiled, not run on hardware
//
#include <hip/hip_runtime.h>
#include <hip/hip_bf16.h>
#include <stdint.h>

typedef __bf16 bf16_t;
typedef __attribute__((ext_vector_type(16))) __bf16 v16bf;
typedef __attribute__((ext_vector_type(8)))  __bf16 v8bf;
typedef __attribute__((ext_vector_type(8)))  float  v8f;

#define B_     8
#define T_     8192
#define D_     512
#define H_     512
#define M_     (B_*T_)      /* 65536 rows of x */
#define NCOMB  1024         /* [Wz | Wh] columns */
#define NCHUNK 32
#define CHUNK  256

/* scratch layout (bytes):
   whi/wlo : 1 MB each   bf16 [1024][512] column-major W (hi/lo split)
   ZT      : 256 MB f32  [65536][1024] (z-logits | th); z slots reused for U
   S, O    : 512 KB each f32 [B][32][H] chunk sums / offsets
   xhi/xlo : 64 MB each  bf16 [65536][512] split of x                     */
#define OFF_WHI ((size_t)0)
#define OFF_WLO ((size_t)(1u<<20))
#define OFF_ZT  ((size_t)(2u<<20))
#define OFF_S   (OFF_ZT + (size_t)M_*NCOMB*4)
#define OFF_O   (OFF_S + (size_t)B_*NCHUNK*H_*4)
#define OFF_XHI (OFF_O + (size_t)B_*NCHUNK*H_*4)
#define OFF_XLO (OFF_XHI + (size_t)M_*D_*2)

// ---------------- K1: transpose + bf16 hi/lo split of [Wz|Wh] -------------
__global__ void k1_split_w(const float* __restrict__ Wz, const float* __restrict__ Wh,
                           bf16_t* __restrict__ whi, bf16_t* __restrict__ wlo) {
  int idx = blockIdx.x * 256 + threadIdx.x;     // 1024*512 elements
  int n = idx >> 9;
  int k = idx & 511;
  float w = (n < 512) ? Wz[(size_t)k * 512 + n] : Wh[(size_t)k * 512 + (n - 512)];
  bf16_t h = (bf16_t)w;
  whi[(size_t)n * 512 + k] = h;
  wlo[(size_t)n * 512 + k] = (bf16_t)(w - (float)h);
}

// ---------------- K1x: bf16 hi/lo split of x (one-time, memory bound) -----
__global__ void k1x_split_x(const float* __restrict__ x,
                            bf16_t* __restrict__ xhi, bf16_t* __restrict__ xlo) {
  size_t i = ((size_t)blockIdx.x * 256 + threadIdx.x) * 8;   // 33.5M elems / 8
  float4 v0 = *(const float4*)(x + i);
  float4 v1 = *(const float4*)(x + i + 4);
  float f[8] = {v0.x, v0.y, v0.z, v0.w, v1.x, v1.y, v1.z, v1.w};
  v8bf h, l;
  #pragma unroll
  for (int e = 0; e < 8; ++e) {
    bf16_t hb = (bf16_t)f[e];
    h[e] = hb;
    l[e] = (bf16_t)(f[e] - (float)hb);
  }
  *(v8bf*)(xhi + i) = h;
  *(v8bf*)(xlo + i) = l;
}

// ---------------- K2: split-bf16 WMMA GEMM: ZT = x @ [Wz|Wh] --------------
// Staging via CDNA5 async global->LDS copies; zero conversion VALU in loop.
__launch_bounds__(256)
__global__ void k2_gemm(const bf16_t* __restrict__ xhi, const bf16_t* __restrict__ xlo,
                        const bf16_t* __restrict__ whi, const bf16_t* __restrict__ wlo,
                        float* __restrict__ ZT) {
  __shared__ __align__(16) bf16_t smhi[2][64 * 32];   // 8 KB (double-buffered)
  __shared__ __align__(16) bf16_t smlo[2][64 * 32];   // 8 KB
  const int tid  = threadIdx.x;
  const int lane = tid & 31;
  const int wave = tid >> 5;
  const int m_w  = wave & 1;        // 2 waves along M
  const int n_w  = wave >> 1;       // 4 waves along N
  const int row0 = blockIdx.y * 64;                 // M block
  const int colW = blockIdx.x * 128 + n_w * 32;     // wave's N base in [0,1024)
  const int half = lane >> 4;
  const int lmod = lane & 15;

  __builtin_prefetch(whi + (size_t)(colW + lmod) * 512, 0, 0);
  __builtin_prefetch(wlo + (size_t)(colW + lmod) * 512, 0, 0);

  // staging geometry: 64 rows x 64 bytes per chunk = 256 x 16B segments
  const int seg = tid & 3;                 // 4 x 16B per row
  const int rr  = tid >> 2;                // 64 rows
  const unsigned growb = (unsigned)(((size_t)(row0 + rr) * 512) * 2);  // row byte base

  // stage chunk 0 asynchronously
  {
    unsigned vo = growb + seg * 16;
    unsigned lh = (unsigned)(uintptr_t)((char*)&smhi[0][0] + rr * 64 + seg * 16);
    unsigned ll = (unsigned)(uintptr_t)((char*)&smlo[0][0] + rr * 64 + seg * 16);
    asm volatile("global_load_async_to_lds_b128 %0, %1, %2"
                 :: "v"(lh), "v"(vo), "s"(xhi) : "memory");
    asm volatile("global_load_async_to_lds_b128 %0, %1, %2"
                 :: "v"(ll), "v"(vo), "s"(xlo) : "memory");
  }
  asm volatile("s_wait_asynccnt 0x0" ::: "memory");
  __syncthreads();

  v8f acc[2][2] = {};
  union Frag { v16bf v; uint4 q[2]; };

  for (int kc = 0; kc < 16; ++kc) {
    const int buf = kc & 1;
    if (kc < 15) {   // async-prefetch next K chunk into the other buffer
      unsigned vo = growb + (unsigned)(kc + 1) * 64 + seg * 16;
      unsigned lh = (unsigned)(uintptr_t)((char*)&smhi[buf ^ 1][0] + rr * 64 + seg * 16);
      unsigned ll = (unsigned)(uintptr_t)((char*)&smlo[buf ^ 1][0] + rr * 64 + seg * 16);
      asm volatile("global_load_async_to_lds_b128 %0, %1, %2"
                   :: "v"(lh), "v"(vo), "s"(xhi) : "memory");
      asm volatile("global_load_async_to_lds_b128 %0, %1, %2"
                   :: "v"(ll), "v"(vo), "s"(xlo) : "memory");
    }

    // B fragments (32x16 bf16): VGPR j holds K = kc*32 + 16*half + 2j,+1
    v16bf bhi[2], blo[2];
    const int krow = kc * 32 + 16 * half;
    #pragma unroll
    for (int nt = 0; nt < 2; ++nt) {
      int n = colW + nt * 16 + lmod;
      bhi[nt] = *(const v16bf*)(whi + (size_t)n * 512 + krow);
      blo[nt] = *(const v16bf*)(wlo + (size_t)n * 512 + krow);
    }

    // A fragments (16x32 bf16, contiguous K per half): pure LDS b128 loads
    v16bf ahi[2], alo[2];
    #pragma unroll
    for (int mt = 0; mt < 2; ++mt) {
      int r = m_w * 32 + mt * 16 + lmod;
      const char* ph = (const char*)&smhi[buf][r * 32];
      const char* pl = (const char*)&smlo[buf][r * 32];
      Frag fh, fl;
      fh.q[0] = *(const uint4*)(ph + 32 * half);        // k-local 16h..16h+7
      fh.q[1] = *(const uint4*)(ph + 16 + 32 * half);   // k-local 16h+8..16h+15
      fl.q[0] = *(const uint4*)(pl + 32 * half);
      fl.q[1] = *(const uint4*)(pl + 16 + 32 * half);
      ahi[mt] = fh.v;
      alo[mt] = fl.v;
    }

    // bf16x3 split product: hi*hi + hi*lo + lo*hi (f32 accumulate)
    #pragma unroll
    for (int mt = 0; mt < 2; ++mt)
      #pragma unroll
      for (int nt = 0; nt < 2; ++nt) {
        acc[mt][nt] = __builtin_amdgcn_wmma_f32_16x16x32_bf16(
            false, ahi[mt], false, bhi[nt], (short)0, acc[mt][nt], false, false);
        acc[mt][nt] = __builtin_amdgcn_wmma_f32_16x16x32_bf16(
            false, ahi[mt], false, blo[nt], (short)0, acc[mt][nt], false, false);
        acc[mt][nt] = __builtin_amdgcn_wmma_f32_16x16x32_bf16(
            false, alo[mt], false, bhi[nt], (short)0, acc[mt][nt], false, false);
      }

    asm volatile("s_wait_asynccnt 0x0" ::: "memory");
    __syncthreads();
  }

  // store C tiles: VGPR r -> row (r + 8*half), col lmod
  #pragma unroll
  for (int mt = 0; mt < 2; ++mt)
    #pragma unroll
    for (int nt = 0; nt < 2; ++nt) {
      int col = colW + nt * 16 + lmod;
      #pragma unroll
      for (int r = 0; r < 8; ++r) {
        int row = row0 + m_w * 32 + mt * 16 + r + 8 * half;
        ZT[(size_t)row * 1024 + col] = acc[mt][nt][r];
      }
    }
}

// ------- K3: bias+sigmoid, prefix over H (in-lane + wave32 shuffle scan) ---
__launch_bounds__(256)
__global__ void k3_row(const float* __restrict__ bz, const float* __restrict__ bh,
                       float* __restrict__ ZT, float* __restrict__ outA) {
  const int lane = threadIdx.x & 31;
  const int wave = threadIdx.x >> 5;
  const int r = blockIdx.x * 8 + wave;            // 65536 rows
  const size_t zb = (size_t)r * 1024 + lane * 16;
  float zl[16], th[16], bzv[16], bhv[16];
  #pragma unroll
  for (int q = 0; q < 4; ++q) {
    *(float4*)&zl[q * 4]  = *(const float4*)(ZT + zb + q * 4);
    *(float4*)&th[q * 4]  = *(const float4*)(ZT + zb + 512 + q * 4);
    *(float4*)&bzv[q * 4] = *(const float4*)(bz + lane * 16 + q * 4);
    *(float4*)&bhv[q * 4] = *(const float4*)(bh + lane * 16 + q * 4);
  }
  float z[16], p[16], run = 0.f;
  #pragma unroll
  for (int i = 0; i < 16; ++i) {
    float s = 1.f / (1.f + __expf(-(zl[i] + bzv[i])));
    z[i] = s;
    run += 1.f - s;
    p[i] = run;
  }
  float tot = run, s = run;
  #pragma unroll
  for (int off = 1; off < 32; off <<= 1) {
    float v = __shfl_up(s, off);
    if (lane >= off) s += v;
  }
  const float excl = s - tot;        // exclusive prefix of lane totals
  float A[16], U[16];
  #pragma unroll
  for (int i = 0; i < 16; ++i) {
    float cs = excl + p[i];          // cumsum over hidden dim (a_star)
    A[i] = cs;
    U[i] = z[i] * (th[i] + bhv[i]) - cs;
  }
  const size_t ob = (size_t)r * 512 + lane * 16;
  #pragma unroll
  for (int q = 0; q < 4; ++q) {
    *(float4*)(outA + ob + q * 4) = *(float4*)&A[q * 4];   // d_out := a_star
    *(float4*)(ZT + zb + q * 4)   = *(float4*)&U[q * 4];   // reuse z slots for U
  }
}

// ---------------- K4: per-chunk time partial sums of U ---------------------
__global__ void k4_chunksum(const float* __restrict__ ZT, float* __restrict__ S) {
  int idx = blockIdx.x * 256 + threadIdx.x;   // B*NCHUNK*H = 131072
  int h = idx & 511;
  int c = (idx >> 9) & 31;
  int b = idx >> 14;
  size_t base = (size_t)(b * T_ + c * CHUNK) * 1024 + h;
  float s = 0.f;
  for (int t = 0; t < CHUNK; ++t) s += ZT[base + (size_t)t * 1024];
  S[idx] = s;                                  // (b*32+c)*512+h == idx
}

// ---------------- K5: exclusive scan of chunk sums, seeded with h0 ---------
__global__ void k5_scan(const float* __restrict__ S, const float* __restrict__ h0,
                        float* __restrict__ O) {
  int idx = blockIdx.x * 256 + threadIdx.x;    // B*H = 4096
  int h = idx & 511;
  int b = idx >> 9;
  float run = h0[b * 512 + h];
  for (int c = 0; c < NCHUNK; ++c) {
    int j = (b * NCHUNK + c) * 512 + h;
    O[j] = run;
    run += S[j];
  }
}

// ---------------- K6: apply scan: out += running prefix; emit h_last -------
__global__ void k6_apply(const float* __restrict__ ZT, const float* __restrict__ O,
                         float* __restrict__ out) {
  int idx = blockIdx.x * 256 + threadIdx.x;    // B*NCHUNK*H
  int h = idx & 511;
  int c = (idx >> 9) & 31;
  int b = idx >> 14;
  float run = O[idx];
  size_t rbase = (size_t)(b * T_ + c * CHUNK);
  float v = 0.f;
  for (int t = 0; t < CHUNK; ++t) {
    size_t r = rbase + t;
    run += ZT[r * 1024 + h];
    v = out[r * 512 + h] + run;                // a_star + (h0 + prefix(U))
    out[r * 512 + h] = v;
  }
  if (c == NCHUNK - 1) out[(size_t)M_ * 512 + b * 512 + h] = v;   // h_last
}

extern "C" void kernel_launch(void* const* d_in, const int* in_sizes, int n_in,
                              void* d_out, int out_size, void* d_ws, size_t ws_size,
                              hipStream_t stream) {
  const float* x  = (const float*)d_in[0];
  const float* h0 = (const float*)d_in[1];
  const float* Wz = (const float*)d_in[2];
  const float* bz = (const float*)d_in[3];
  const float* Wh = (const float*)d_in[4];
  const float* bh = (const float*)d_in[5];
  float* out = (float*)d_out;
  char* ws = (char*)d_ws;
  bf16_t* whi = (bf16_t*)(ws + OFF_WHI);
  bf16_t* wlo = (bf16_t*)(ws + OFF_WLO);
  float*  ZT  = (float*)(ws + OFF_ZT);
  float*  S   = (float*)(ws + OFF_S);
  float*  O   = (float*)(ws + OFF_O);
  bf16_t* xhi = (bf16_t*)(ws + OFF_XHI);
  bf16_t* xlo = (bf16_t*)(ws + OFF_XLO);

  k1_split_w<<<dim3(2048), dim3(256), 0, stream>>>(Wz, Wh, whi, wlo);
  k1x_split_x<<<dim3(16384), dim3(256), 0, stream>>>(x, xhi, xlo);
  k2_gemm<<<dim3(8, 1024), dim3(256), 0, stream>>>(xhi, xlo, whi, wlo, ZT);
  k3_row<<<dim3(8192), dim3(256), 0, stream>>>(bz, bh, ZT, out);
  k4_chunksum<<<dim3(512), dim3(256), 0, stream>>>(ZT, S);
  k5_scan<<<dim3(16), dim3(256), 0, stream>>>(S, h0, O);
  k6_apply<<<dim3(512), dim3(256), 0, stream>>>(ZT, O, out);
}
